// NodeLayer_58205396796108
// MI455X (gfx1250) — matrix operand order repaired and checked by reading, not compiled, not run on hardware
//
#include <hip/hip_runtime.h>

typedef __attribute__((ext_vector_type(16))) _Float16 v16h;
typedef __attribute__((ext_vector_type(8)))  float    v8f;

#define LDS_BYTES 313856  // 128K W1 + 128K W2 + 16K xA + 32K hA + 2K b1 + 0.5K b2

// One workgroup (8 waves, 256 thr) integrates 64 rows for all 20 RK4 steps.
// Weights: f16 in LDS, pre-packed per-lane in WMMA B-fragment order.
// State x / RK4 accumulator: registers, WMMA C/D layout.
// x and h round-trip through LDS staging pre-arranged in A-fragment order.
// Register tiling: GEMM1 2Mx4N per wave (B frags reused x2),
//                  GEMM2 2Mx2N per wave (A and B frags reused x2).
__global__ __launch_bounds__(256, 1)
void node_rk4_kernel(const float* __restrict__ x0,
                     const float* __restrict__ W1,
                     const float* __restrict__ b1,
                     const float* __restrict__ W2,
                     const float* __restrict__ b2,
                     float* __restrict__ out)
{
  constexpr float DTT = 0.1f;
  extern __shared__ char smem[];
  _Float16* w1p = (_Float16*)(smem);            // [32 nt][4 ks][32 lane][16 half]
  _Float16* w2p = (_Float16*)(smem + 131072);   // [8 nt][16 ks][32 lane][16 half]
  _Float16* xa  = (_Float16*)(smem + 262144);   // [4 m][4 ks] A-frag blocks (x)
  _Float16* ha  = (_Float16*)(smem + 278528);   // [4 m][8 ks] A-frag blocks (h half)
  float*    b1s = (float*)   (smem + 311296);
  float*    b2s = (float*)   (smem + 313344);

  const int tid = threadIdx.x;
  const int wv  = tid >> 5;
  const int ln  = tid & 31;
  const int lh  = ln & 15;
  const int hig = ln >> 4;
  const int mp  = wv >> 2;        // M-tile pair: M-tiles {2mp, 2mp+1}
  const int ntq = wv & 3;         // N quarter

  // ---- pack weights f32 -> f16 into B-fragment order: K=8*(ln>=16)+s (s<8), 16+8*(ln>=16)+s-8 (s>=8), N=ln%16
  for (int i = tid; i < 128 * 512; i += 256) {
    int k = i >> 9, n = i & 511, kk = k & 31;
    w1p[((n >> 4) * 4 + (k >> 5)) * 512 + ((n & 15) + (((kk >> 3) & 1) << 4)) * 16 + ((kk & 7) + ((kk >> 4) << 3))]
        = (_Float16)W1[i];
  }
  for (int i = tid; i < 512 * 128; i += 256) {
    int k = i >> 7, n = i & 127, kk = k & 31;
    w2p[((n >> 4) * 16 + (k >> 5)) * 512 + ((n & 15) + (((kk >> 3) & 1) << 4)) * 16 + ((kk & 7) + ((kk >> 4) << 3))]
        = (_Float16)W2[i];
  }
  for (int i = tid; i < 512; i += 256) b1s[i] = b1[i];
  if (tid < 128) b2s[tid] = b2[tid];

  // ---- register state (C/D layout: row = v + 8*hig + 16*mt, col = nt*16 + lh)
  float xs[2][2][8], acc[2][2][8], b2v[2];
  int   xoff[2][2];
  long  rb[2];
  #pragma unroll
  for (int mi = 0; mi < 2; ++mi)
    rb[mi] = (long)blockIdx.x * 64 + (2 * mp + mi) * 16 + 8 * hig;
  #pragma unroll
  for (int nj = 0; nj < 2; ++nj) {
    const int col  = (2 * ntq + nj) * 16 + lh;
    const int kk   = col & 31;
    const int slot = ((kk & 7) & ~1) + ((kk >> 4) << 3) + (kk & 1);  // A K-map
    const int g8   = (kk >> 3) & 1;
    #pragma unroll
    for (int mi = 0; mi < 2; ++mi) {
      xoff[mi][nj] = ((2 * mp + mi) * 4 + (col >> 5)) * 512 + (8 * hig + 16 * g8) * 16 + slot;
      #pragma unroll
      for (int v = 0; v < 8; ++v) xs[mi][nj][v] = x0[(rb[mi] + v) * 128 + col];
    }
  }
  __syncthreads();
  #pragma unroll
  for (int nj = 0; nj < 2; ++nj) b2v[nj] = b2s[(2 * ntq + nj) * 16 + lh];
  #pragma unroll
  for (int mi = 0; mi < 2; ++mi)
    #pragma unroll
    for (int nj = 0; nj < 2; ++nj)
      #pragma unroll
      for (int v = 0; v < 8; ++v) xa[xoff[mi][nj] + v * 16] = (_Float16)xs[mi][nj][v];

  const v8f vzero = {};

  #pragma unroll 1
  for (int e = 0; e < 80; ++e) {     // 20 RK4 steps x 4 stages
    const int s = e & 3;
    __syncthreads();                 // xa ready; previous ha readers done

    v16h a1[2][4];                   // x A-fragments for both owned M-tiles
    #pragma unroll
    for (int mi = 0; mi < 2; ++mi)
      #pragma unroll
      for (int ks = 0; ks < 4; ++ks)
        a1[mi][ks] = *(const v16h*)(xa + ((2 * mp + mi) * 4 + ks) * 512 + ln * 16);

    v8f D4[2][2];
    #pragma unroll
    for (int mi = 0; mi < 2; ++mi)
      #pragma unroll
      for (int nj = 0; nj < 2; ++nj) D4[mi][nj] = vzero;

    #pragma unroll 1
    for (int hh = 0; hh < 2; ++hh) {
      // GEMM1 half: h[:, hh*256:+256] = relu(x @ W1 + b1) -> ha (A-frag order)
      #pragma unroll
      for (int jj = 0; jj < 4; ++jj) {
        const int ntg = (hh << 4) + (ntq << 2) + jj;
        v8f d[2];
        d[0] = vzero; d[1] = vzero;
        #pragma unroll
        for (int ks = 0; ks < 4; ++ks) {
          v16h bf = *(const v16h*)(w1p + (ntg * 4 + ks) * 512 + ln * 16);
          #pragma unroll
          for (int mi = 0; mi < 2; ++mi)
            d[mi] = __builtin_amdgcn_wmma_f32_16x16x32_f16(false, a1[mi][ks], false, bf,
                                                           (short)0, d[mi], false, false);
        }
        const int col  = ntg * 16 + lh;
        const float bb = b1s[col];
        const int khh  = col & 255;
        const int kk   = khh & 31;
        const int slot = ((kk & 7) & ~1) + ((kk >> 4) << 3) + (kk & 1);
        const int g8   = (kk >> 3) & 1;
        #pragma unroll
        for (int mi = 0; mi < 2; ++mi) {
          const int hb = ((2 * mp + mi) * 8 + (khh >> 5)) * 512 + (8 * hig + 16 * g8) * 16 + slot;
          #pragma unroll
          for (int v = 0; v < 8; ++v) {
            float hv = d[mi][v] + bb;
            hv = hv > 0.f ? hv : 0.f;
            ha[hb + v * 16] = (_Float16)hv;
          }
        }
      }
      __syncthreads();               // ha visible
      // GEMM2 partial: accumulate K slice [hh*256, +256)
      #pragma unroll
      for (int ks = 0; ks < 8; ++ks) {
        v16h af[2];
        #pragma unroll
        for (int mi = 0; mi < 2; ++mi)
          af[mi] = *(const v16h*)(ha + ((2 * mp + mi) * 8 + ks) * 512 + ln * 16);
        #pragma unroll
        for (int nj = 0; nj < 2; ++nj) {
          v16h bf = *(const v16h*)(w2p + ((2 * ntq + nj) * 16 + (hh * 8 + ks)) * 512 + ln * 16);
          #pragma unroll
          for (int mi = 0; mi < 2; ++mi)
            D4[mi][nj] = __builtin_amdgcn_wmma_f32_16x16x32_f16(false, af[mi], false, bf,
                                                               (short)0, D4[mi][nj], false, false);
        }
      }
      if (hh == 0) __syncthreads();  // ha readers done before overwrite
    }

    // RK4 stage update in f32, write next WMMA input to xa
    #pragma unroll
    for (int mi = 0; mi < 2; ++mi) {
      #pragma unroll
      for (int nj = 0; nj < 2; ++nj) {
        #pragma unroll
        for (int v = 0; v < 8; ++v) {
          const float kv = D4[mi][nj][v] + b2v[nj];
          float xc;
          if (s == 0)      { acc[mi][nj][v]  = kv;       xc = xs[mi][nj][v] + 0.5f * DTT * kv; }
          else if (s == 1) { acc[mi][nj][v] += 2.f * kv; xc = xs[mi][nj][v] + 0.5f * DTT * kv; }
          else if (s == 2) { acc[mi][nj][v] += 2.f * kv; xc = xs[mi][nj][v] + DTT * kv; }
          else { xs[mi][nj][v] += (DTT / 6.f) * (acc[mi][nj][v] + kv); xc = xs[mi][nj][v]; }
          xa[xoff[mi][nj] + v * 16] = (_Float16)xc;
        }
      }
    }
  }

  #pragma unroll
  for (int mi = 0; mi < 2; ++mi)
    #pragma unroll
    for (int nj = 0; nj < 2; ++nj) {
      const int col = (2 * ntq + nj) * 16 + lh;
      #pragma unroll
      for (int v = 0; v < 8; ++v) out[(rb[mi] + v) * 128 + col] = xs[mi][nj][v];
    }
}

extern "C" void kernel_launch(void* const* d_in, const int* in_sizes, int n_in,
                              void* d_out, int out_size, void* d_ws, size_t ws_size,
                              hipStream_t stream) {
  (void)in_sizes; (void)n_in; (void)out_size; (void)d_ws; (void)ws_size;
  const float* x0 = (const float*)d_in[0];
  const float* W1 = (const float*)d_in[1];
  const float* b1 = (const float*)d_in[2];
  const float* W2 = (const float*)d_in[3];
  const float* b2 = (const float*)d_in[4];
  float* out = (float*)d_out;
  hipFuncSetAttribute((const void*)node_rk4_kernel,
                      hipFuncAttributeMaxDynamicSharedMemorySize, LDS_BYTES);
  node_rk4_kernel<<<dim3(65536 / 64), dim3(256), LDS_BYTES, stream>>>(
      x0, W1, b1, W2, b2, out);
}
